// BaseAttention_18399639896816
// MI455X (gfx1250) — compile-verified, matrix operand order checked by reading, not run on hardware
//
#include <hip/hip_runtime.h>
#include <hip/hip_bf16.h>

// MHA: B=2, S=2048, H=1024, NH=16, HD=64. bf16 WMMA (f32 accum) everywhere.
// Register-blocked: GEMMs compute 64x64 C per wave (16 WMMAs / 8 frag loads
// per k-step); attention handles 32 q-rows per wave (K/V frags reused 2x).
// __launch_bounds__(256,1) raises the VGPR budget (wave32 allows up to 1024)
// so the 4x4 accumulator block lives entirely in registers (no scratch spill).

typedef __attribute__((ext_vector_type(16))) __bf16 v16bf;
typedef __attribute__((ext_vector_type(8)))  __bf16 v8bf;
typedef __attribute__((ext_vector_type(8)))  float  v8f;

#define WMMA_BF16(a, b, c) \
  __builtin_amdgcn_wmma_f32_16x16x32_bf16(false, (a), false, (b), (short)0, (c), false, false)

static constexpr int Bb    = 2;
static constexpr int Ss    = 2048;
static constexpr int Hh    = 1024;
static constexpr int NH    = 16;
static constexpr int HD    = 64;
static constexpr int Mrows = Bb * Ss;          // 4096
static constexpr float SCALE = 0.125f;         // 64^-0.5

// ---------------------------------------------------------------------------
// Fragment loader: 16-bit A/B operand, row-major tile with leading dim `ld`.
// Lane l (<16) holds row l, K = {0..7, 16..23}; lane l+16 holds row l,
// K = {8..15, 24..31}.  Two 16-byte contiguous loads per lane.
// ---------------------------------------------------------------------------
__device__ __forceinline__ v16bf load_frag(const __bf16* tile, int ld, int lane) {
  const int l    = lane & 15;
  const int half = lane >> 4;
  const __bf16* p = tile + l * ld + half * 8;
  v8bf lo = *(const v8bf*)(p);
  v8bf hi = *(const v8bf*)(p + 16);
  return __builtin_shufflevector(lo, hi, 0,1,2,3,4,5,6,7,8,9,10,11,12,13,14,15);
}

__device__ __forceinline__ float rmax16(float v) {
#pragma unroll
  for (int off = 1; off < 16; off <<= 1) v = fmaxf(v, __shfl_xor(v, off, 32));
  return v;
}
__device__ __forceinline__ float rsum16(float v) {
#pragma unroll
  for (int off = 1; off < 16; off <<= 1) v += __shfl_xor(v, off, 32);
  return v;
}

// ---------------------------------------------------------------------------
// f32 -> bf16 cast
// ---------------------------------------------------------------------------
__global__ void __launch_bounds__(256)
cvt_bf16_kernel(const float* __restrict__ in, __bf16* __restrict__ out, int n) {
  int i = blockIdx.x * blockDim.x + threadIdx.x;
  if (i < n) out[i] = (__bf16)in[i];
}

// ---------------------------------------------------------------------------
// QKV projection: qkv = hidden @ w_qkv^T  (NT GEMM, M=4096, N=3072, K=1024).
// One wave computes a 64x64 C macro-tile (4x4 WMMA fragments).  Scatters into
// Q[B,NH,S,HD], K[B,NH,S,HD], Vt[B,NH,HD,S] (pre-transposed for PV operand).
// A 64-wide column tile never straddles a q/k/v boundary or a head boundary.
// ---------------------------------------------------------------------------
__global__ void __launch_bounds__(256, 1)
gemm_qkv_kernel(const __bf16* __restrict__ hid,
                const __bf16* __restrict__ wqkv,
                __bf16* __restrict__ q, __bf16* __restrict__ k,
                __bf16* __restrict__ vt) {
  const int lane = threadIdx.x & 31;
  const int wId  = (blockIdx.x * blockDim.x + threadIdx.x) >> 5;
  const int mT = wId & 63;         // 64 row macro-tiles   (M/64)
  const int nT = wId >> 6;         // 48 col macro-tiles   (N/64)
  const __bf16* A  = hid  + mT * 64 * Hh;
  const __bf16* Bm = wqkv + nT * 64 * Hh;

  v8f c[4][4] = {};
#pragma unroll 1
  for (int kk = 0; kk < Hh; kk += 32) {
    v16bf a[4], b[4];
#pragma unroll
    for (int i = 0; i < 4; ++i) a[i] = load_frag(A  + (i * 16) * Hh + kk, Hh, lane);
#pragma unroll
    for (int j = 0; j < 4; ++j) b[j] = load_frag(Bm + (j * 16) * Hh + kk, Hh, lane);
#pragma unroll
    for (int i = 0; i < 4; ++i)
#pragma unroll
      for (int j = 0; j < 4; ++j)
        c[i][j] = WMMA_BF16(a[i], b[j], c[i][j]);
  }

  const int cidx  = lane & 15;
  const int n0    = nT * 64;
  const int which = n0 >> 10;              // 0=q 1=k 2=v (constant per wave)
  const int h     = (n0 & 1023) >> 6;      // head (constant per wave)
  const int rowHalf = (lane >> 4) * 8;
#pragma unroll
  for (int i = 0; i < 4; ++i) {
#pragma unroll
    for (int j = 0; j < 4; ++j) {
      const int d = j * 16 + cidx;         // 0..63 within head
#pragma unroll
      for (int r = 0; r < 8; ++r) {
        const int row = mT * 64 + i * 16 + rowHalf + r;
        const int b_  = row >> 11;         // /2048
        const int s   = row & 2047;
        const __bf16 val = (__bf16)c[i][j][r];
        if (which == 0)      q [((b_ * NH + h) * Ss + s) * HD + d] = val;
        else if (which == 1) k [((b_ * NH + h) * Ss + s) * HD + d] = val;
        else                 vt[((b_ * NH + h) * HD + d) * Ss + s] = val;
      }
    }
  }
}

// ---------------------------------------------------------------------------
// Fused flash attention.  One wave handles 32 query rows (2 row tiles) of one
// (b,h).  Keys in chunks of 32: per row tile, scores via 4 WMMAs on shared K
// fragments, online softmax in C-fragment layout, P staged through LDS to
// re-layout as an A fragment, PV via 4 WMMAs on shared V fragments.
// ---------------------------------------------------------------------------
__global__ void __launch_bounds__(256, 1)
attn_kernel(const __bf16* __restrict__ q,
            const __bf16* __restrict__ kk,
            const __bf16* __restrict__ vt,
            const int* __restrict__ amask,
            __bf16* __restrict__ attn_out) {
  __shared__ __bf16 pLds[8][16 * 40];    // per-wave 16x32 P tile, ld=40 (pad)

  const int lane   = threadIdx.x & 31;
  const int wLocal = threadIdx.x >> 5;
  const int wGlob  = blockIdx.x * 8 + wLocal;
  const int qT = wGlob & 63;             // S/32 = 64 query macro-tiles
  const int bh = wGlob >> 6;
  const int h  = bh & 15;
  const int b  = bh >> 4;

  const __bf16* qh = q  + ((b * NH + h) * Ss + qT * 32) * HD;
  const __bf16* kh = kk + ((b * NH + h) * Ss) * HD;
  const __bf16* vh = vt + ((b * NH + h) * HD) * Ss;
  const int* mrow  = amask + b * Ss;

  v16bf qf[2][2];
#pragma unroll
  for (int t = 0; t < 2; ++t) {
    qf[t][0] = load_frag(qh + t * 16 * HD,      HD, lane);
    qf[t][1] = load_frag(qh + t * 16 * HD + 32, HD, lane);
  }

  v8f o[2][4] = {};
  float mst[2][8], lst[2][8];
#pragma unroll
  for (int t = 0; t < 2; ++t)
#pragma unroll
    for (int r = 0; r < 8; ++r) { mst[t][r] = -1e30f; lst[t][r] = 0.0f; }

  __bf16* pw = pLds[wLocal];
  const int cidx = lane & 15;
  const int rowB = (lane >> 4) * 8;

  for (int kb = 0; kb < Ss; kb += 32) {
    // ---- shared K fragments for this chunk ----
    v16bf bk[2][2];
#pragma unroll
    for (int j = 0; j < 2; ++j) {
      bk[j][0] = load_frag(kh + (kb + j * 16) * HD,      HD, lane);
      bk[j][1] = load_frag(kh + (kb + j * 16) * HD + 32, HD, lane);
    }
    const float madd0 = (mrow[kb + cidx]      == 0) ? -1e30f : 0.0f;
    const float madd1 = (mrow[kb + 16 + cidx] == 0) ? -1e30f : 0.0f;

    // ---- shared V fragments for this chunk ----
    v16bf vf[4];
#pragma unroll
    for (int n = 0; n < 4; ++n)
      vf[n] = load_frag(vh + (n * 16) * Ss + kb, Ss, lane);

#pragma unroll
    for (int t = 0; t < 2; ++t) {
      // scores: 16q x 32k
      v8f s0 = {}, s1 = {};
      s0 = WMMA_BF16(qf[t][0], bk[0][0], s0);
      s0 = WMMA_BF16(qf[t][1], bk[0][1], s0);
      s1 = WMMA_BF16(qf[t][0], bk[1][0], s1);
      s1 = WMMA_BF16(qf[t][1], bk[1][1], s1);

      // online softmax (row stats replicated across each 16-lane half)
#pragma unroll
      for (int r = 0; r < 8; ++r) {
        float a0 = s0[r] * SCALE + madd0;
        float a1 = s1[r] * SCALE + madd1;
        float mx    = rmax16(fmaxf(a0, a1));
        float mnew  = fmaxf(mst[t][r], mx);
        float alpha = __expf(mst[t][r] - mnew);
        mst[t][r] = mnew;
        float p0 = __expf(a0 - mnew);
        float p1 = __expf(a1 - mnew);
        lst[t][r] = lst[t][r] * alpha + rsum16(p0 + p1);
        o[t][0][r] *= alpha; o[t][1][r] *= alpha;
        o[t][2][r] *= alpha; o[t][3][r] *= alpha;
        s0[r] = p0; s1[r] = p1;
      }

      // stage P (C layout) into LDS, reload as A fragment, accumulate PV.
      // LDS ops are in-order within a wave; compiler inserts the dscnt wait.
#pragma unroll
      for (int r = 0; r < 8; ++r) {
        pw[(rowB + r) * 40 + cidx]      = (__bf16)s0[r];
        pw[(rowB + r) * 40 + 16 + cidx] = (__bf16)s1[r];
      }
      const v16bf pf = load_frag(pw, 40, lane);
#pragma unroll
      for (int n = 0; n < 4; ++n)
        o[t][n] = WMMA_BF16(pf, vf[n], o[t][n]);
    }
  }

  // ---- normalize and write [B*S, H] bf16 for the output GEMM ----
#pragma unroll
  for (int t = 0; t < 2; ++t) {
#pragma unroll
    for (int r = 0; r < 8; ++r) {
      const float inv = 1.0f / lst[t][r];
      const int qrow = qT * 32 + t * 16 + rowB + r;
      __bf16* dst = attn_out + (b * Ss + qrow) * Hh + h * HD;
      dst[ 0 + cidx] = (__bf16)(o[t][0][r] * inv);
      dst[16 + cidx] = (__bf16)(o[t][1][r] * inv);
      dst[32 + cidx] = (__bf16)(o[t][2][r] * inv);
      dst[48 + cidx] = (__bf16)(o[t][3][r] * inv);
    }
  }
}

// ---------------------------------------------------------------------------
// Output projection: out = attn @ w_out^T  (M=4096, N=1024, K=1024), f32 out.
// 64x64 C macro-tile per wave.
// ---------------------------------------------------------------------------
__global__ void __launch_bounds__(256, 1)
gemm_out_kernel(const __bf16* __restrict__ attn,
                const __bf16* __restrict__ wout,
                float* __restrict__ out) {
  const int lane = threadIdx.x & 31;
  const int wId  = (blockIdx.x * blockDim.x + threadIdx.x) >> 5;
  const int mT = wId & 63;         // 64 row macro-tiles
  const int nT = wId >> 6;         // 16 col macro-tiles
  const __bf16* A  = attn + mT * 64 * Hh;
  const __bf16* Bm = wout + nT * 64 * Hh;

  v8f c[4][4] = {};
#pragma unroll 1
  for (int kk = 0; kk < Hh; kk += 32) {
    v16bf a[4], b[4];
#pragma unroll
    for (int i = 0; i < 4; ++i) a[i] = load_frag(A  + (i * 16) * Hh + kk, Hh, lane);
#pragma unroll
    for (int j = 0; j < 4; ++j) b[j] = load_frag(Bm + (j * 16) * Hh + kk, Hh, lane);
#pragma unroll
    for (int i = 0; i < 4; ++i)
#pragma unroll
      for (int j = 0; j < 4; ++j)
        c[i][j] = WMMA_BF16(a[i], b[j], c[i][j]);
  }

  const int cidx = lane & 15;
  const int rowHalf = (lane >> 4) * 8;
#pragma unroll
  for (int i = 0; i < 4; ++i)
#pragma unroll
    for (int j = 0; j < 4; ++j) {
      const int col = nT * 64 + j * 16 + cidx;
#pragma unroll
      for (int r = 0; r < 8; ++r)
        out[(mT * 64 + i * 16 + rowHalf + r) * Hh + col] = c[i][j][r];
    }
}

// ---------------------------------------------------------------------------
extern "C" void kernel_launch(void* const* d_in, const int* in_sizes, int n_in,
                              void* d_out, int out_size, void* d_ws, size_t ws_size,
                              hipStream_t stream) {
  const float* hidden = (const float*)d_in[0];   // [2,2048,1024] f32
  const int*   amask  = (const int*)d_in[1];     // [2,2048] i32
  const float* w_qkv  = (const float*)d_in[2];   // [3072,1024] f32
  const float* w_out  = (const float*)d_in[3];   // [1024,1024] f32
  float* out = (float*)d_out;                    // [2,2048,1024] f32

  // workspace layout (bytes)
  char* ws = (char*)d_ws;
  __bf16* hid_bf  = (__bf16*)(ws);                              // 8 MB
  __bf16* wqkv_bf = (__bf16*)(ws + (8u  << 20));                // 6 MB
  __bf16* wout_bf = (__bf16*)(ws + (14u << 20));                // 2 MB
  __bf16* q_bf    = (__bf16*)(ws + (16u << 20));                // 8 MB
  __bf16* k_bf    = (__bf16*)(ws + (24u << 20));                // 8 MB
  __bf16* vt_bf   = (__bf16*)(ws + (32u << 20));                // 8 MB (transposed)
  __bf16* attn_bf = (__bf16*)(ws + (40u << 20));                // 8 MB

  const int nHid  = Mrows * Hh;        // 4194304
  const int nWqkv = 3 * Hh * Hh;       // 3145728
  const int nWout = Hh * Hh;           // 1048576

  cvt_bf16_kernel<<<(nHid  + 255) / 256, 256, 0, stream>>>(hidden, hid_bf,  nHid);
  cvt_bf16_kernel<<<(nWqkv + 255) / 256, 256, 0, stream>>>(w_qkv,  wqkv_bf, nWqkv);
  cvt_bf16_kernel<<<(nWout + 255) / 256, 256, 0, stream>>>(w_out,  wout_bf, nWout);

  // QKV GEMM: (4096/64)*(3072/64) = 3072 wave-tiles, 8 waves/block
  gemm_qkv_kernel<<<3072 / 8, 256, 0, stream>>>(hid_bf, wqkv_bf, q_bf, k_bf, vt_bf);

  // flash attention: B*NH*(S/32) = 2048 wave-tiles
  attn_kernel<<<2048 / 8, 256, 0, stream>>>(q_bf, k_bf, vt_bf, amask, attn_bf);

  // output GEMM: (4096/64)*(1024/64) = 1024 wave-tiles
  gemm_out_kernel<<<1024 / 8, 256, 0, stream>>>(attn_bf, wout_bf, out);

  (void)in_sizes; (void)n_in; (void)out_size; (void)ws_size;
}
